// EngineWithScatter_22651657519432
// MI455X (gfx1250) — matrix operand order repaired and checked by reading, not compiled, not run on hardware
//
#include <hip/hip_runtime.h>
#include <hip/hip_bf16.h>
#include <math.h>

// ---------------------------------------------------------------------------
// MoE FFN (top-2 of 8 experts), N=8192 tokens, C=1024, H=4096, on gfx1250.
// Sparse routing + bf16 WMMA (v_wmma_f32_16x16x32_bf16) grouped GEMMs.
// Staging uses CDNA5 async global->LDS copies (ASYNCcnt) via inline asm.
// ---------------------------------------------------------------------------

typedef __attribute__((ext_vector_type(16))) __bf16 v16bf;
typedef __attribute__((ext_vector_type(8)))  float  v8f;

constexpr int NB   = 4;
constexpr int NT   = 2048;
constexpr int NC   = 1024;   // model dim
constexpr int NH   = 4096;   // hidden dim
constexpr int NE   = 8;      // experts
constexpr int NTOK = NB * NT;        // 8192 tokens
constexpr int NSLOT = NTOK * 2;      // top-2 -> 16384 slot rows total

// -------------------- workspace layout (bytes) --------------------
constexpr size_t OFF_XB   = 0;                                   // bf16 x       (N*C)
constexpr size_t SZ_XB    = (size_t)NTOK * NC * 2;
constexpr size_t OFF_W1B  = OFF_XB + SZ_XB;                      // bf16 W1      (E*C*H)
constexpr size_t SZ_W1B   = (size_t)NE * NC * NH * 2;
constexpr size_t OFF_W2B  = OFF_W1B + SZ_W1B;                    // bf16 W2      (E*H*C)
constexpr size_t SZ_W2B   = (size_t)NE * NH * NC * 2;
constexpr size_t OFF_H    = OFF_W2B + SZ_W2B;                    // bf16 h       (2N*H)
constexpr size_t SZ_H     = (size_t)NSLOT * NH * 2;
constexpr size_t OFF_CNT  = OFF_H + SZ_H;                        // int counts[E]
constexpr size_t OFF_OFFS = OFF_CNT + 256;                       // int offsets[E]
constexpr size_t OFF_CUR  = OFF_OFFS + 256;                      // int cursor[E]
constexpr size_t OFF_TE   = OFF_CUR + 256;                       // int tokExp[N*2]
constexpr size_t OFF_TG   = OFF_TE + (size_t)NTOK * 2 * 4;       // f32 tokGate[N*2]
constexpr size_t OFF_S2T  = OFF_TG + (size_t)NTOK * 2 * 4;       // int slot2tok[2N]
constexpr size_t OFF_SG   = OFF_S2T + (size_t)NSLOT * 4;         // f32 slotGate[2N]

__device__ __forceinline__ unsigned short f32_to_bf16(float f) {
  unsigned int u = __builtin_bit_cast(unsigned int, f);
  unsigned int r = (u + 0x7FFFu + ((u >> 16) & 1u)) >> 16;  // round-to-nearest-even
  return (unsigned short)r;
}

// Publish the LDS base address into asm once per kernel. This makes the
// __shared__ array's address escape, so the compiler must assume the
// "memory"-clobbering async-copy asm below writes it -- otherwise LLVM sees a
// never-stored LDS global and folds the WMMA fragment loads to undef.
__device__ __forceinline__ void publish_lds(void* p) {
  asm volatile("" : : "v"((unsigned long long)(uintptr_t)p) : "memory");
}

// CDNA5 async copy: 16 bytes per lane, global -> LDS, tracked by ASYNCcnt.
// lds_off is the byte offset into the workgroup LDS allocation.
__device__ __forceinline__ void async_copy_b128(unsigned lds_off, const void* gptr) {
  asm volatile("global_load_async_to_lds_b128 %0, %1, off"
               :
               : "v"(lds_off), "v"((unsigned long long)(uintptr_t)gptr)
               : "memory");
}
__device__ __forceinline__ void wait_async0() {
  asm volatile("s_wait_asynccnt 0x0" ::: "memory");
}

// -------------------- fp32 -> bf16 conversion --------------------
__global__ void k_f32_to_bf16(const float* __restrict__ src,
                              unsigned short* __restrict__ dst, int n) {
  int i = blockIdx.x * blockDim.x + threadIdx.x;
  int stride = gridDim.x * blockDim.x;
  for (; i < n; i += stride) dst[i] = f32_to_bf16(src[i]);
}

// -------------------- router: one wave32 per token --------------------
__global__ void k_router(const float* __restrict__ x, const float* __restrict__ Wr,
                         int* __restrict__ counts, int* __restrict__ tokExp,
                         float* __restrict__ tokGate) {
  const int wid = threadIdx.x >> 5, lane = threadIdx.x & 31;
  const int tok = blockIdx.x * 8 + wid;
  if (tok >= NTOK) return;
  float acc[NE];
#pragma unroll
  for (int e = 0; e < NE; e++) acc[e] = 0.f;
  const float* xr = x + (size_t)tok * NC;
  for (int c = lane; c < NC; c += 32) {
    float xv = xr[c];
    const float* wr = Wr + (size_t)c * NE;
#pragma unroll
    for (int e = 0; e < NE; e++) acc[e] += xv * wr[e];
  }
#pragma unroll
  for (int e = 0; e < NE; e++)
    for (int off = 16; off > 0; off >>= 1) acc[e] += __shfl_down(acc[e], off, 32);
  if (lane == 0) {
    int i0 = 0; float b0 = acc[0];
    for (int e = 1; e < NE; e++) if (acc[e] > b0) { b0 = acc[e]; i0 = e; }
    int i1 = -1; float b1 = -1e30f;
    for (int e = 0; e < NE; e++) if (e != i0 && acc[e] > b1) { b1 = acc[e]; i1 = e; }
    float e1 = __expf(b1 - b0);
    float s  = 1.f + e1;                 // softmax over the selected pair
    tokExp[tok * 2 + 0] = i0; tokExp[tok * 2 + 1] = i1;
    tokGate[tok * 2 + 0] = 1.f / s; tokGate[tok * 2 + 1] = e1 / s;
    atomicAdd(&counts[i0], 1);
    atomicAdd(&counts[i1], 1);
  }
}

// -------------------- prefix (E=8, trivial) --------------------
__global__ void k_prefix(const int* __restrict__ counts, int* __restrict__ offs,
                         int* __restrict__ cursor) {
  if (blockIdx.x == 0 && threadIdx.x == 0) {
    int acc = 0;
    for (int e = 0; e < NE; e++) { offs[e] = acc; acc += counts[e]; cursor[e] = 0; }
  }
}

// -------------------- slot assignment --------------------
__global__ void k_assign(const int* __restrict__ tokExp, const float* __restrict__ tokGate,
                         const int* __restrict__ offs, int* __restrict__ cursor,
                         int* __restrict__ s2t, float* __restrict__ sg) {
  int t = blockIdx.x * blockDim.x + threadIdx.x;
  if (t >= NTOK) return;
#pragma unroll
  for (int j = 0; j < 2; j++) {
    int e = tokExp[t * 2 + j];
    int pos = atomicAdd(&cursor[e], 1);
    int slot = offs[e] + pos;
    s2t[slot] = t;
    sg[slot]  = tokGate[t * 2 + j];
  }
}

// ---------------------------------------------------------------------------
// Grouped GEMM tiles: 128(M) x 128(N), K step 32. 256 threads = 8 waves in a
// 2x4 grid; each wave owns 64x32 = 4x2 WMMA 16x16 accumulators.
// LDS: sA = bytes [0, 8K)   (128 rows x 32 bf16)
//      sB = bytes [8K, 16K) (32 rows x 128 bf16)
// Overhang M rows load clamped (garbage) data: row m of A only feeds row m of
// the output tile, and the epilogue masks r >= rows, so no zero-fill needed.
// ---------------------------------------------------------------------------
constexpr unsigned LDS_B_BYTES = 128 * 32 * 2;   // 8192

// Up-proj: h[slot, :] = gelu( x[tok(slot), :] @ W1[e] ), stored bf16.
__global__ void __launch_bounds__(256) k_moe_up(
    const unsigned short* __restrict__ xb, const unsigned short* __restrict__ w1b,
    const int* __restrict__ counts, const int* __restrict__ offs,
    const int* __restrict__ s2t, unsigned short* __restrict__ hbuf) {
  const int e  = blockIdx.x / (NTOK / 128);
  const int tm = blockIdx.x % (NTOK / 128);
  const int rows = counts[e];
  if (tm * 128 >= rows) return;                 // sparse early-exit (uniform)
  const int h0   = blockIdx.y * 128;
  const int base = offs[e];

  __shared__ unsigned short smem[128 * 32 + 32 * 128];   // 16 KB, offset 0 in LDS
  unsigned short* sA = smem;
  unsigned short* sB = smem + 128 * 32;
  publish_lds(smem);                            // see comment above

  const int tid = threadIdx.x;
  const int lane = tid & 31;
  const int wid  = tid >> 5;
  const int wr = wid >> 2, wc = wid & 3;        // wave (row, col) in 2x4
  v8f acc[4][2] = {};

  // staging roles
  const int ar = tid >> 1, ah = tid & 1;        // A: row 0..127, 16-ushort half
  const int sr = tm * 128 + ar;
  const int atok = s2t[base + ((sr < rows) ? sr : (rows - 1))];  // clamped gather
  const int br = tid >> 3, bs = tid & 7;        // B: row 0..31, 16-ushort seg
  const unsigned ldsA = (unsigned)(ar * 32 + ah * 16) * 2;       // byte offsets
  const unsigned ldsB = LDS_B_BYTES + (unsigned)(br * 128 + bs * 16) * 2;

  for (int k0 = 0; k0 < NC; k0 += 32) {
    __syncthreads();
    // stage A (gathered x rows) and B (W1[e] rows): direct global->LDS async
    {
      const unsigned short* ga = xb + (size_t)atok * NC + k0 + ah * 16;
      async_copy_b128(ldsA,      ga);
      async_copy_b128(ldsA + 16, ga + 8);
      const unsigned short* gb = w1b + ((size_t)e * NC + k0 + br) * NH + h0 + bs * 16;
      async_copy_b128(ldsB,      gb);
      async_copy_b128(ldsB + 16, gb + 8);
      if (k0 + 32 < NC) __builtin_prefetch(gb + 32 * NH, 0, 1);  // global_prefetch_b8
    }
    wait_async0();
    __syncthreads();

    // B fragments: lane = N col (mod 16), lanes>=16 hold K 16..31, 2 K / vgpr
    union { v16bf v; unsigned int u[8]; } bf[2];
#pragma unroll
    for (int tn = 0; tn < 2; tn++) {
      const int n  = wc * 32 + tn * 16 + (lane & 15);
      const int kb = (lane >> 4) * 16;
#pragma unroll
      for (int v = 0; v < 8; v++) {
        unsigned int lo = sB[(kb + 2 * v)     * 128 + n];
        unsigned int hi = sB[(kb + 2 * v + 1) * 128 + n];
        bf[tn].u[v] = lo | (hi << 16);
      }
    }
    // A fragments: lane = M row (mod 16), lanes>=16 hold K 16..31
#pragma unroll
    for (int ti = 0; ti < 4; ti++) {
      union { v16bf v; unsigned int u[8]; } af;
      const int m  = wr * 64 + ti * 16 + (lane & 15);
      const int kb = (lane >> 4) * 16;
#pragma unroll
      for (int v = 0; v < 8; v++) {
        int kp = kb + 2 * v;
        af.u[v] = *reinterpret_cast<const unsigned int*>(&sA[m * 32 + kp]);
      }
#pragma unroll
      for (int tn = 0; tn < 2; tn++)
        acc[ti][tn] = __builtin_amdgcn_wmma_f32_16x16x32_bf16(
            false, af.v, false, bf[tn].v, (short)0, acc[ti][tn], false, false);
    }
  }

  // epilogue: tanh-GELU, bf16 store. C/D layout: vgpr v -> M=v(+8 hi lanes), N=lane&15
#pragma unroll
  for (int ti = 0; ti < 4; ti++)
#pragma unroll
    for (int tn = 0; tn < 2; tn++)
#pragma unroll
      for (int v = 0; v < 8; v++) {
        int mi = v + 8 * (lane >> 4);
        int r  = tm * 128 + wr * 64 + ti * 16 + mi;
        if (r < rows) {
          int col = h0 + wc * 32 + tn * 16 + (lane & 15);
          float xv = acc[ti][tn][v];
          float g  = 0.5f * xv *
                     (1.f + tanhf(0.7978845608028654f * (xv + 0.044715f * xv * xv * xv)));
          hbuf[(size_t)(base + r) * NH + col] = f32_to_bf16(g);
        }
      }
}

// Down-proj: out[tok(slot), :] += gate(slot) * ( h[slot, :] @ W2[e] )
__global__ void __launch_bounds__(256) k_moe_down(
    const unsigned short* __restrict__ hbuf, const unsigned short* __restrict__ w2b,
    const int* __restrict__ counts, const int* __restrict__ offs,
    const int* __restrict__ s2t, const float* __restrict__ sg,
    float* __restrict__ out) {
  const int e  = blockIdx.x / (NTOK / 128);
  const int tm = blockIdx.x % (NTOK / 128);
  const int rows = counts[e];
  if (tm * 128 >= rows) return;
  const int n0   = blockIdx.y * 128;            // over C
  const int base = offs[e];

  __shared__ unsigned short smem[128 * 32 + 32 * 128];
  unsigned short* sA = smem;
  unsigned short* sB = smem + 128 * 32;
  publish_lds(smem);

  const int tid = threadIdx.x;
  const int lane = tid & 31;
  const int wid  = tid >> 5;
  const int wr = wid >> 2, wc = wid & 3;
  v8f acc[4][2] = {};

  const int ar = tid >> 1, ah = tid & 1;
  const int sr = tm * 128 + ar;
  const int arow = base + ((sr < rows) ? sr : (rows - 1));       // clamped
  const int br = tid >> 3, bs = tid & 7;
  const unsigned ldsA = (unsigned)(ar * 32 + ah * 16) * 2;
  const unsigned ldsB = LDS_B_BYTES + (unsigned)(br * 128 + bs * 16) * 2;

  for (int k0 = 0; k0 < NH; k0 += 32) {
    __syncthreads();
    {
      const unsigned short* ga = hbuf + (size_t)arow * NH + k0 + ah * 16;
      async_copy_b128(ldsA,      ga);
      async_copy_b128(ldsA + 16, ga + 8);
      const unsigned short* gb = w2b + ((size_t)e * NH + k0 + br) * NC + n0 + bs * 16;
      async_copy_b128(ldsB,      gb);
      async_copy_b128(ldsB + 16, gb + 8);
      if (k0 + 32 < NH) __builtin_prefetch(gb + 32 * NC, 0, 1);
    }
    wait_async0();
    __syncthreads();

    union { v16bf v; unsigned int u[8]; } bf[2];
#pragma unroll
    for (int tn = 0; tn < 2; tn++) {
      const int n  = wc * 32 + tn * 16 + (lane & 15);
      const int kb = (lane >> 4) * 16;
#pragma unroll
      for (int v = 0; v < 8; v++) {
        unsigned int lo = sB[(kb + 2 * v)     * 128 + n];
        unsigned int hi = sB[(kb + 2 * v + 1) * 128 + n];
        bf[tn].u[v] = lo | (hi << 16);
      }
    }
#pragma unroll
    for (int ti = 0; ti < 4; ti++) {
      union { v16bf v; unsigned int u[8]; } af;
      const int m  = wr * 64 + ti * 16 + (lane & 15);
      const int kb = (lane >> 4) * 16;
#pragma unroll
      for (int v = 0; v < 8; v++) {
        int kp = kb + 2 * v;
        af.u[v] = *reinterpret_cast<const unsigned int*>(&sA[m * 32 + kp]);
      }
#pragma unroll
      for (int tn = 0; tn < 2; tn++)
        acc[ti][tn] = __builtin_amdgcn_wmma_f32_16x16x32_bf16(
            false, af.v, false, bf[tn].v, (short)0, acc[ti][tn], false, false);
    }
  }

  // epilogue: gate-scaled scatter-add into fp32 output (global_atomic_add_f32)
#pragma unroll
  for (int ti = 0; ti < 4; ti++)
#pragma unroll
    for (int tn = 0; tn < 2; tn++)
#pragma unroll
      for (int v = 0; v < 8; v++) {
        int mi = v + 8 * (lane >> 4);
        int r  = tm * 128 + wr * 64 + ti * 16 + mi;
        if (r < rows) {
          int slot = base + r;
          int tok  = s2t[slot];
          float g  = sg[slot];
          int col  = n0 + wc * 32 + tn * 16 + (lane & 15);
          unsafeAtomicAdd(&out[(size_t)tok * NC + col], g * acc[ti][tn][v]);
        }
      }
}

// ---------------------------------------------------------------------------
extern "C" void kernel_launch(void* const* d_in, const int* in_sizes, int n_in,
                              void* d_out, int out_size, void* d_ws, size_t ws_size,
                              hipStream_t stream) {
  const float* x  = (const float*)d_in[0];
  const float* Wr = (const float*)d_in[1];
  const float* W1 = (const float*)d_in[2];
  const float* W2 = (const float*)d_in[3];
  float* out = (float*)d_out;

  char* ws = (char*)d_ws;
  unsigned short* xb   = (unsigned short*)(ws + OFF_XB);
  unsigned short* w1b  = (unsigned short*)(ws + OFF_W1B);
  unsigned short* w2b  = (unsigned short*)(ws + OFF_W2B);
  unsigned short* hbuf = (unsigned short*)(ws + OFF_H);
  int*   counts  = (int*)(ws + OFF_CNT);
  int*   offs    = (int*)(ws + OFF_OFFS);
  int*   cursor  = (int*)(ws + OFF_CUR);
  int*   tokExp  = (int*)(ws + OFF_TE);
  float* tokGate = (float*)(ws + OFF_TG);
  int*   s2t     = (int*)(ws + OFF_S2T);
  float* sg      = (float*)(ws + OFF_SG);

  hipMemsetAsync(ws + OFF_CNT, 0, 768, stream);                       // counts/offs/cursor
  hipMemsetAsync(d_out, 0, (size_t)NTOK * NC * sizeof(float), stream);

  k_f32_to_bf16<<<4096, 256, 0, stream>>>(x,  xb,  NTOK * NC);
  k_f32_to_bf16<<<8192, 256, 0, stream>>>(W1, w1b, NE * NC * NH);
  k_f32_to_bf16<<<8192, 256, 0, stream>>>(W2, w2b, NE * NH * NC);

  k_router<<<NTOK / 8, 256, 0, stream>>>(x, Wr, counts, tokExp, tokGate);
  k_prefix<<<1, 1, 0, stream>>>(counts, offs, cursor);
  k_assign<<<NTOK / 256, 256, 0, stream>>>(tokExp, tokGate, offs, cursor, s2t, sg);

  dim3 gup(NE * (NTOK / 128), NH / 128);   // (512, 32), most M-tiles early-exit
  k_moe_up<<<gup, 256, 0, stream>>>(xb, w1b, counts, offs, s2t, hbuf);

  dim3 gdn(NE * (NTOK / 128), NC / 128);   // (512, 8)
  k_moe_down<<<gdn, 256, 0, stream>>>(hbuf, w2b, counts, offs, s2t, sg, out);
}
